// AttentionBlock_15676630630630
// MI455X (gfx1250) — compile-verified, matrix operand order checked by reading, not run on hardware
//
#include <hip/hip_runtime.h>

typedef __attribute__((ext_vector_type(16))) _Float16 v16h;
typedef __attribute__((ext_vector_type(8)))  _Float16 v8h;
typedef __attribute__((ext_vector_type(4)))  _Float16 v4h;
typedef __attribute__((ext_vector_type(8)))  float    v8f;
typedef __attribute__((ext_vector_type(4)))  float    v4f;
typedef __attribute__((ext_vector_type(4)))  int      v4i;

// ---------------------------------------------------------------------------
// CDNA5 async global->LDS copy (ASYNCcnt-tracked), with sync fallback.
// ---------------------------------------------------------------------------
#if __has_builtin(__builtin_amdgcn_global_load_async_to_lds_b128) && \
    __has_builtin(__builtin_amdgcn_s_wait_asynccnt)
#define HAS_ASYNC_LDS 1
#else
#define HAS_ASYNC_LDS 0
#endif

__device__ __forceinline__ void cp_b128_to_lds(const _Float16* src, _Float16* dstLds)
{
#if HAS_ASYNC_LDS
    __builtin_amdgcn_global_load_async_to_lds_b128(
        (__attribute__((address_space(1))) v4i*)(void*)src,
        (__attribute__((address_space(3))) v4i*)(void*)dstLds, 0, 0);
#else
    *(v8h*)dstLds = *(const v8h*)src;
#endif
}

template <int N>
__device__ __forceinline__ void cp_wait()
{
#if HAS_ASYNC_LDS
    __builtin_amdgcn_s_wait_asynccnt(N);
#endif
}

// ---------------------------------------------------------------------------
// GroupNorm: x[b][c][n] f32 -> hn[b][n][c] f16 (token-major)
// grid = (32 groups, 8 batches), block = 256.
// Pass 2: one token per lane; the 16 group-channels are contiguous in hn,
// so each lane emits two coalesced 16B stores.
// ---------------------------------------------------------------------------
__global__ __launch_bounds__(256) void gn_kernel(
    const float* __restrict__ x, const float* __restrict__ gamma,
    const float* __restrict__ beta, _Float16* __restrict__ hn)
{
    const int g   = blockIdx.x;
    const int b   = blockIdx.y;
    const int tid = threadIdx.x;
    const float* xb = x + ((size_t)b * 512 + (size_t)g * 16) * 4096;

    float s = 0.f, s2 = 0.f;
    for (int i = tid; i < 16 * 4096; i += 256) {
        float v = xb[i];
        s += v; s2 += v * v;
    }
    __shared__ float r1[256], r2[256];
    r1[tid] = s; r2[tid] = s2;
    __syncthreads();
    for (int off = 128; off > 0; off >>= 1) {
        if (tid < off) { r1[tid] += r1[tid + off]; r2[tid] += r2[tid + off]; }
        __syncthreads();
    }
    const float mean = r1[0] * (1.0f / 65536.0f);
    const float var  = r2[0] * (1.0f / 65536.0f) - mean * mean;
    const float rstd = rsqrtf(var + 1e-5f);

    float ga[16], be[16];
#pragma unroll
    for (int cl = 0; cl < 16; cl++) {
        ga[cl] = gamma[g * 16 + cl] * rstd;
        be[cl] = beta[g * 16 + cl];
    }

    _Float16* hb = hn + (size_t)b * 4096 * 512 + g * 16;
    for (int n = tid; n < 4096; n += 256) {
        v8h lo, hi;
#pragma unroll
        for (int cl = 0; cl < 8; cl++)
            lo[cl] = (_Float16)((xb[cl * 4096 + n] - mean) * ga[cl] + be[cl]);
#pragma unroll
        for (int cl = 8; cl < 16; cl++)
            hi[cl - 8] = (_Float16)((xb[cl * 4096 + n] - mean) * ga[cl] + be[cl]);
        _Float16* dst = hb + (size_t)n * 512;
        *(v8h*)dst = lo;
        *(v8h*)(dst + 8) = hi;
    }
}

// ---------------------------------------------------------------------------
// f32 -> f16 weight conversion
// ---------------------------------------------------------------------------
__global__ __launch_bounds__(256) void cvt_kernel(
    const float* __restrict__ in, _Float16* __restrict__ out, int n)
{
    for (int i = blockIdx.x * 256 + threadIdx.x; i < n; i += gridDim.x * 256)
        out[i] = (_Float16)in[i];
}

// ---------------------------------------------------------------------------
// Row softmax, in place on f16 scores. grid = 4096 rows, block = 256, N = 4096
// ---------------------------------------------------------------------------
__global__ __launch_bounds__(256) void softmax_kernel(_Float16* __restrict__ S)
{
    _Float16* p = S + (size_t)blockIdx.x * 4096;
    const int tid = threadIdx.x;
    float v[16];
    float mx = -3.0e30f;
#pragma unroll
    for (int j = 0; j < 16; j++) {
        v[j] = (float)p[tid + j * 256];
        mx = fmaxf(mx, v[j]);
    }
    __shared__ float red[256];
    red[tid] = mx;
    __syncthreads();
    for (int off = 128; off > 0; off >>= 1) {
        if (tid < off) red[tid] = fmaxf(red[tid], red[tid + off]);
        __syncthreads();
    }
    mx = red[0];
    __syncthreads();
    float s = 0.f;
#pragma unroll
    for (int j = 0; j < 16; j++) { v[j] = __expf(v[j] - mx); s += v[j]; }
    red[tid] = s;
    __syncthreads();
    for (int off = 128; off > 0; off >>= 1) {
        if (tid < off) red[tid] += red[tid + off];
        __syncthreads();
    }
    const float inv = 1.0f / red[0];
#pragma unroll
    for (int j = 0; j < 16; j++) p[tid + j * 256] = (_Float16)(v[j] * inv);
}

// ---------------------------------------------------------------------------
// WMMA GEMM: C[i,j] = scale * sum_k A[i,k]*B[j,k] (+bias[j]) (+resid)
// Block tile 128x128 (8 waves 4x2, wave 32x64 = 2x4 WMMA tiles), K-step 32,
// double-buffered LDS via async global->LDS. OUT_TRANS uses an LDS-staged
// transpose (overlaid on the dead k-loop buffers) for coalesced stores.
// ---------------------------------------------------------------------------
constexpr int TM = 128, TN = 128, TK = 32, LPAD = 40;
constexpr int SMEM_BYTES = 2 * (TM + TN) * LPAD * (int)sizeof(_Float16);  // 40960

template <bool OUT_F16, bool OUT_TRANS, bool RESID>
__global__ __launch_bounds__(256) void gemm_tn(
    const _Float16* __restrict__ A, const _Float16* __restrict__ B,
    const float* __restrict__ bias, void* __restrict__ outv,
    const float* __restrict__ resid,
    int K, int lda, int ldb, int ldo, float scale,
    long long aBatch, long long oBatch, long long rBatch)
{
    __shared__ __align__(16) char smem[SMEM_BYTES];
    _Float16 (*lA)[TM][LPAD] = (_Float16(*)[TM][LPAD])smem;
    _Float16 (*lB)[TN][LPAD] =
        (_Float16(*)[TN][LPAD])(smem + 2 * TM * LPAD * sizeof(_Float16));

    const int tid  = threadIdx.x;
    const int lane = tid & 31;
    const int wave = tid >> 5;
    const int wm   = wave & 3;              // wave row 0..3  -> rows wm*32
    const int wn   = wave >> 2;             // wave col 0..1  -> cols wn*64
    const int fr   = lane & 15;             // fragment row/col
    const int hs   = (lane & 16) ? 8 : 0;   // K half-select (halves)

    const int rowBase = blockIdx.x * TM;
    const int colBase = blockIdx.y * TN;

    A += (size_t)blockIdx.z * (size_t)aBatch;

    const int sr = tid >> 2;                // 0..63
    const int sc = (tid & 3) << 3;          // 0,8,16,24

    const _Float16* gA0 = A + (unsigned)(rowBase + sr) * (unsigned)lda + sc;
    const _Float16* gA1 = A + (unsigned)(rowBase + sr + 64) * (unsigned)lda + sc;
    const _Float16* gB0 = B + (unsigned)(colBase + sr) * (unsigned)ldb + sc;
    const _Float16* gB1 = B + (unsigned)(colBase + sr + 64) * (unsigned)ldb + sc;

    v8f acc[2][4] = {};

    const int KT = K / TK;

    cp_b128_to_lds(gA0, &lA[0][sr][sc]);
    cp_b128_to_lds(gA1, &lA[0][sr + 64][sc]);
    cp_b128_to_lds(gB0, &lB[0][sr][sc]);
    cp_b128_to_lds(gB1, &lB[0][sr + 64][sc]);

    for (int kt = 0; kt < KT; ++kt) {
        const int cur = kt & 1;
        if (kt + 1 < KT) {
            const int k1 = (kt + 1) * TK;
            cp_b128_to_lds(gA0 + k1, &lA[cur ^ 1][sr][sc]);
            cp_b128_to_lds(gA1 + k1, &lA[cur ^ 1][sr + 64][sc]);
            cp_b128_to_lds(gB0 + k1, &lB[cur ^ 1][sr][sc]);
            cp_b128_to_lds(gB1 + k1, &lB[cur ^ 1][sr + 64][sc]);
            cp_wait<4>();
        } else {
            cp_wait<0>();
        }
        __syncthreads();

        v16h af[2], bf[4];
#pragma unroll
        for (int i = 0; i < 2; i++) {
            const _Float16* pr = &lA[cur][wm * 32 + i * 16 + fr][0];
            v8h lo = *(const v8h*)(pr + hs);
            v8h hi = *(const v8h*)(pr + 16 + hs);
#pragma unroll
            for (int t = 0; t < 8; t++) { af[i][t] = lo[t]; af[i][t + 8] = hi[t]; }
        }
#pragma unroll
        for (int i = 0; i < 4; i++) {
            const _Float16* pc = &lB[cur][wn * 64 + i * 16 + fr][0];
            v8h lo = *(const v8h*)(pc + hs);
            v8h hi = *(const v8h*)(pc + 16 + hs);
#pragma unroll
            for (int t = 0; t < 8; t++) { bf[i][t] = lo[t]; bf[i][t + 8] = hi[t]; }
        }
#pragma unroll
        for (int mi = 0; mi < 2; mi++)
#pragma unroll
            for (int ni = 0; ni < 4; ni++)
                acc[mi][ni] = __builtin_amdgcn_wmma_f32_16x16x32_f16(
                    false, af[mi], false, bf[ni], (short)0, acc[mi][ni],
                    false, false);
        __syncthreads();
    }

    const int mofs = (lane & 16) ? 8 : 0;
    const size_t zO = (size_t)blockIdx.z * (size_t)oBatch;
    const size_t zR = (size_t)blockIdx.z * (size_t)rBatch;

    if (OUT_TRANS) {
        // two-phase transpose through LDS (overlaid on dead k-loop buffers)
        float (*tt)[132] = (float(*)[132])smem;    // [64 cols][128 rows] f32
#pragma unroll
        for (int p = 0; p < 2; p++) {
            __syncthreads();
            if (wn == p) {
#pragma unroll
                for (int ni = 0; ni < 4; ni++)
#pragma unroll
                    for (int mi = 0; mi < 2; mi++) {
                        const int m0 = wm * 32 + mi * 16 + mofs;
#pragma unroll
                        for (int j = 0; j < 8; j++)
                            tt[ni * 16 + fr][m0 + j] = acc[mi][ni][j];
                    }
            }
            __syncthreads();
#pragma unroll
            for (int it = 0; it < 8; it++) {
                const int idx  = it * 256 + tid;
                const int ch   = idx >> 5;            // col within half: 0..63
                const int row0 = (idx & 31) << 2;     // 0..124
                const int col  = colBase + p * 64 + ch;
                const float bb = bias ? bias[col] : 0.0f;
                const unsigned base =
                    (unsigned)col * (unsigned)ldo + (unsigned)(rowBase + row0);
                v4f v = *(const v4f*)&tt[ch][row0];
                v4f r = {};
                if (RESID) r = *(const v4f*)(resid + zR + base);
                if (OUT_F16) {
                    v4h h;
#pragma unroll
                    for (int q = 0; q < 4; q++) {
                        float vv = v[q] * scale + bb;
                        if (RESID) vv += r[q];
                        h[q] = (_Float16)vv;
                    }
                    *(v4h*)((_Float16*)outv + zO + base) = h;
                } else {
                    v4f o;
#pragma unroll
                    for (int q = 0; q < 4; q++) {
                        float vv = v[q] * scale + bb;
                        if (RESID) vv += r[q];
                        o[q] = vv;
                    }
                    *(v4f*)((float*)outv + zO + base) = o;
                }
            }
        }
    } else {
#pragma unroll
        for (int ni = 0; ni < 4; ni++) {
            const int col = colBase + wn * 64 + ni * 16 + fr;
            const float bb = bias ? bias[col] : 0.0f;
#pragma unroll
            for (int mi = 0; mi < 2; mi++) {
                const int row0 = rowBase + wm * 32 + mi * 16 + mofs;
#pragma unroll
                for (int j = 0; j < 8; j++) {
                    float vsum = acc[mi][ni][j] * scale + bb;
                    const unsigned idx =
                        (unsigned)(row0 + j) * (unsigned)ldo + (unsigned)col;
                    if (RESID) vsum += resid[zR + idx];
                    if (OUT_F16)
                        ((_Float16*)outv)[zO + idx] = (_Float16)vsum;
                    else
                        ((float*)outv)[zO + idx] = vsum;
                }
            }
        }
    }
}

// ---------------------------------------------------------------------------
// Host-side orchestration
// ---------------------------------------------------------------------------
extern "C" void kernel_launch(void* const* d_in, const int* in_sizes, int n_in,
                              void* d_out, int out_size, void* d_ws, size_t ws_size,
                              hipStream_t stream)
{
    const float* x        = (const float*)d_in[0];
    const float* gn_scale = (const float*)d_in[1];
    const float* gn_bias  = (const float*)d_in[2];
    const float* wq = (const float*)d_in[3];
    const float* bq = (const float*)d_in[4];
    const float* wk = (const float*)d_in[5];
    const float* bk = (const float*)d_in[6];
    const float* wv = (const float*)d_in[7];
    const float* bv = (const float*)d_in[8];
    const float* wp = (const float*)d_in[9];
    const float* bp = (const float*)d_in[10];
    float* out = (float*)d_out;

    const int B = 8, C = 512, N = 4096;
    const size_t ACT = (size_t)B * N * C * sizeof(_Float16);   // 32 MiB
    const size_t SCO = (size_t)N * N * sizeof(_Float16);       // 32 MiB
    const size_t WGT = (size_t)C * C * sizeof(_Float16);       // 512 KiB

    char* w = (char*)d_ws;
    _Float16* hn   = (_Float16*)(w);
    _Float16* Qb   = (_Float16*)(w + ACT);
    _Float16* Kb   = (_Float16*)(w + 2 * ACT);
    _Float16* Vt   = (_Float16*)(w + 3 * ACT);      // channel-major [b][c][n]
    _Float16* attn = (_Float16*)(w + 4 * ACT);      // token-major [b][n][c]
    _Float16* Sb   = (_Float16*)(w + 5 * ACT);      // one batch of scores
    _Float16* wqh  = (_Float16*)(w + 5 * ACT + SCO);
    _Float16* wkh  = (_Float16*)(w + 5 * ACT + SCO + WGT);
    _Float16* wvh  = (_Float16*)(w + 5 * ACT + SCO + 2 * WGT);
    _Float16* wph  = (_Float16*)(w + 5 * ACT + SCO + 3 * WGT);

    cvt_kernel<<<256, 256, 0, stream>>>(wq, wqh, C * C);
    cvt_kernel<<<256, 256, 0, stream>>>(wk, wkh, C * C);
    cvt_kernel<<<256, 256, 0, stream>>>(wv, wvh, C * C);
    cvt_kernel<<<256, 256, 0, stream>>>(wp, wph, C * C);

    gn_kernel<<<dim3(32, 8), 256, 0, stream>>>(x, gn_scale, gn_bias, hn);

    const long long nc = (long long)N * C;

    gemm_tn<true, false, false><<<dim3(B * N / TM, C / TN, 1), 256, 0, stream>>>(
        hn, wqh, bq, Qb, nullptr, C, C, C, C, 1.0f, 0, 0, 0);
    gemm_tn<true, false, false><<<dim3(B * N / TM, C / TN, 1), 256, 0, stream>>>(
        hn, wkh, bk, Kb, nullptr, C, C, C, C, 1.0f, 0, 0, 0);
    gemm_tn<true, true, false><<<dim3(N / TM, C / TN, B), 256, 0, stream>>>(
        hn, wvh, bv, Vt, nullptr, C, C, C, N, 1.0f, nc, nc, 0);

    const float scal = 0.044194173824159216f;   // 1/sqrt(512)
    for (int b = 0; b < B; b++) {
        gemm_tn<true, false, false><<<dim3(N / TM, N / TN, 1), 256, 0, stream>>>(
            Qb + (size_t)b * nc, Kb + (size_t)b * nc, nullptr, Sb, nullptr,
            C, C, C, N, scal, 0, 0, 0);
        softmax_kernel<<<N, 256, 0, stream>>>(Sb);
        gemm_tn<true, false, false><<<dim3(N / TM, C / TN, 1), 256, 0, stream>>>(
            Sb, Vt + (size_t)b * nc, nullptr, attn + (size_t)b * nc, nullptr,
            N, N, N, C, 1.0f, 0, 0, 0);
    }

    gemm_tn<false, true, true><<<dim3(N / TM, C / TN, B), 256, 0, stream>>>(
        attn, wph, bp, out, x, C, C, C, N, 1.0f, nc, nc, nc);
}
